// GAT_71597104824719
// MI455X (gfx1250) — matrix operand order repaired and checked by reading, not compiled, not run on hardware
//
#include <hip/hip_runtime.h>

// ---------------------------------------------------------------------------
// GAT (2-layer, PyG-style) for MI455X / gfx1250.
// Dense GEMMs -> v_wmma_f32_16x16x32_bf16 (fp32 accumulate), weights staged
// into LDS once per workgroup via the Tensor Data Mover (tensor_load_to_lds,
// TDM pad feature gives a bank-conflict-free 272B row stride), B fragments
// then come from ds_load_b128.
// Edge softmax/aggregation -> L2-resident float atomics (192MB L2 holds all
// node-feature buffers), scores recomputed per pass instead of spilling E*H.
// ---------------------------------------------------------------------------

typedef __attribute__((ext_vector_type(16))) __bf16        v16bf;
typedef __attribute__((ext_vector_type(8)))  float         v8f;
typedef __attribute__((ext_vector_type(4)))  unsigned int  v4u;
typedef __attribute__((ext_vector_type(8)))  int           v8i;
typedef __attribute__((ext_vector_type(4)))  int           v4i;

#define NEG_ATT 0.2f
#define NEG_ACT 0.01f

#if defined(__gfx1250__) && __has_builtin(__builtin_amdgcn_tensor_load_to_lds) && \
    __has_builtin(__builtin_amdgcn_s_wait_tensorcnt)
#define USE_TDM 1
#else
#define USE_TDM 0
#endif

// LDS row stride for staged weights: 128 bf16 (256B) + 16B pad = 136 shorts.
#define WROW 136

// ---- bf16 helpers (stored as unsigned short) ------------------------------
__device__ __forceinline__ unsigned short f2bf(float f) {
    unsigned int u = __float_as_uint(f);
    u += 0x7fffu + ((u >> 16) & 1u);      // round-to-nearest-even
    return (unsigned short)(u >> 16);
}
__device__ __forceinline__ float bf2f(unsigned short h) {
    return __uint_as_float(((unsigned int)h) << 16);
}

// ---- order-preserving int encoding of float (for atomicMax) ---------------
__device__ __forceinline__ int fkey(float f) {
    int i = __float_as_int(f);
    return i >= 0 ? i : (i ^ 0x7fffffff);
}
__device__ __forceinline__ float fval(int k) {
    int i = k >= 0 ? k : (k ^ 0x7fffffff);
    return __int_as_float(i);
}

// ---- WMMA fragments (contiguous k = 16*hi + s in both A and B, wave32) --------------------------
union Frag16 { v16bf v; unsigned int u[8]; };

// A: 16x32 bf16 from global. lane m=lane&15, hi=lane>>4.
//   v0..7 : K = kk + 2v + 16*hi
__device__ __forceinline__ void load_a(const unsigned short* base, int row0,
                                       int ldk, int kk, unsigned lane, Frag16& f) {
    int m  = lane & 15;
    int hi = (lane >> 4) & 1;
    const unsigned short* rp = base + (size_t)(row0 + m) * ldk;
#pragma unroll
    for (int v = 0; v < 8; ++v) {
        int k = kk + 2 * v + 16 * hi;
        f.u[v] = *(const unsigned int*)(rp + k);
    }
}

// B: 32x16 bf16 from LDS-staged transposed weights sW[n][k] (row stride WROW).
//   lane n=lane&15, hi=lane>>4 ; K = kk + 2v + 16*hi  -> two ds_load_b128
__device__ __forceinline__ void load_b_lds(const unsigned short* sW, int col0,
                                           int kk, unsigned lane, Frag16& f) {
    int n  = lane & 15;
    int hi = (lane >> 4) & 1;
    const unsigned int* rp =
        (const unsigned int*)(sW + (size_t)(col0 + n) * WROW + kk + 16 * hi);
#pragma unroll
    for (int v = 0; v < 8; ++v) f.u[v] = rp[v];
}

// Stage Wt[NOUT][128] bf16 -> LDS with 272B padded rows.
template <int NOUT>
__device__ __forceinline__ void stage_weights(const unsigned short* wt,
                                              unsigned short* sW) {
#if USE_TDM
    if (threadIdx.x < 32) {               // one TDM descriptor per workgroup
        unsigned lds_base = (unsigned)(size_t)(&sW[0]);
        unsigned long long ga = (unsigned long long)(size_t)wt;
        // D# group 0: count=1, lds_addr, global_addr[56:0], type=2
        v4u g0 = { 1u,
                   lds_base,
                   (unsigned)(ga & 0xffffffffu),
                   (unsigned)((ga >> 32) & 0x01ffffffu) | (2u << 30) };
        // D# group 1: data_size=2B, pad_enable, pad_interval=64dw (one row),
        // pad_amount=4dw (16B); tensor 128 x NOUT, tile 128 x NOUT, stride 128.
        v8i g1;
        g1[0] = (1 << 16) | (1 << 20) | (5 << 22) | (3 << 25);
        g1[1] = (128 << 16);                       // tensor_dim0[15:0]
        g1[2] = (int)((unsigned)(NOUT & 0xffff) << 16); // tensor_dim1[15:0]
        g1[3] = (128 << 16);                       // tile_dim0
        g1[4] = (NOUT & 0xffff);                   // tile_dim1 (tile_dim2=0)
        g1[5] = 128;                               // tensor_dim0_stride[31:0]
        g1[6] = 0;
        g1[7] = 0;
        v4i z4 = {0, 0, 0, 0};
#if __clang_major__ >= 23
        v8i z8 = {0, 0, 0, 0, 0, 0, 0, 0};
        __builtin_amdgcn_tensor_load_to_lds(g0, g1, z4, z4, z8, 0);
#else
        __builtin_amdgcn_tensor_load_to_lds(g0, g1, z4, z4, 0);
#endif
        __builtin_amdgcn_s_wait_tensorcnt(0);
    }
#else
    for (int idx = threadIdx.x; idx < NOUT * 128; idx += 256) {
        int n = idx >> 7, k = idx & 127;
        sW[n * WROW + k] = wt[n * 128 + k];
    }
#endif
    __syncthreads();
}

// ---- fill / cast kernels --------------------------------------------------
__global__ void k_fill_f32(float* p, float v, long long n) {
    long long i = (long long)blockIdx.x * 256 + threadIdx.x;
    if (i < n) p[i] = v;
}
__global__ void k_fill_i32(int* p, int v, long long n) {
    long long i = (long long)blockIdx.x * 256 + threadIdx.x;
    if (i < n) p[i] = v;
}
__global__ void k_cast_bf16(const float* in, unsigned short* out, long long n) {
    long long i = (long long)blockIdx.x * 256 + threadIdx.x;
    if (i < n) out[i] = f2bf(in[i]);
}
// W[k][n] (KxNout, row-major) -> Wt[n][k] bf16
__global__ void k_cast_w_t(const float* W, unsigned short* Wt, int K, int Nout) {
    int i = blockIdx.x * 256 + threadIdx.x;
    if (i >= K * Nout) return;
    int n = i % Nout, k = i / Nout;
    Wt[(size_t)n * K + k] = f2bf(W[(size_t)k * Nout + n]);
}

// ---- GEMM: out[M,Nout] = in_bf16[M,128] @ Wt_bf16[Nout,128]^T -------------
// 8 waves/block, wave -> one 16-row tile, loops Nout/16 col tiles, K=128.
template <int NOUT>
__global__ void k_gemm_wmma(const unsigned short* __restrict__ inb,
                            const unsigned short* __restrict__ wt,
                            float* __restrict__ out, int ntiles) {
    __shared__ unsigned short sW[NOUT * WROW];
    stage_weights<NOUT>(wt, sW);          // TDM (or fallback) + barrier

    int wave = threadIdx.x >> 5;
    unsigned lane = threadIdx.x & 31;
    int t = blockIdx.x * 8 + wave;
    if (t >= ntiles) return;              // wave-uniform, after the barrier
    const int row0 = t * 16;
    const int NCT = NOUT / 16;

    v8f acc[NCT];
#pragma unroll
    for (int i = 0; i < NCT; ++i) acc[i] = (v8f){0.f,0.f,0.f,0.f,0.f,0.f,0.f,0.f};

#pragma unroll
    for (int kk = 0; kk < 128; kk += 32) {
        Frag16 a; load_a(inb, row0, 128, kk, lane, a);
#pragma unroll
        for (int ct = 0; ct < NCT; ++ct) {
            Frag16 b; load_b_lds(sW, ct * 16, kk, lane, b);
            acc[ct] = __builtin_amdgcn_wmma_f32_16x16x32_bf16(
                false, a.v, false, b.v, (short)0, acc[ct], false, false);
        }
    }
    int n = lane & 15, hi = (lane >> 4) & 1;
#pragma unroll
    for (int ct = 0; ct < NCT; ++ct)
#pragma unroll
        for (int r = 0; r < 8; ++r)
            out[(size_t)(row0 + r + 8 * hi) * NOUT + ct * 16 + n] = acc[ct][r];
}

// scalar tail for rows not covered by 16-row tiles (N=100000 divides evenly,
// kept for generality)
__global__ void k_gemm_tail(const unsigned short* inb, const unsigned short* wt,
                            float* out, int rowStart, int rows, int Nout) {
    int i = blockIdx.x * 256 + threadIdx.x;
    if (i >= rows * Nout) return;
    int r = rowStart + i / Nout, n = i % Nout;
    float s = 0.f;
    for (int k = 0; k < 128; ++k)
        s += bf2f(inb[(size_t)r * 128 + k]) * bf2f(wt[(size_t)n * 128 + k]);
    out[(size_t)r * Nout + n] = s;
}

// ---- per-node attention logits -------------------------------------------
// layer1: one thread per (node, head); h[N,128], a[8,16]
__global__ void k_alpha1(const float* __restrict__ h, const float* __restrict__ a_s,
                         const float* __restrict__ a_d, float* as_, float* ad_,
                         int N) {
    int i = blockIdx.x * 256 + threadIdx.x;
    if (i >= N * 8) return;
    int node = i >> 3, head = i & 7;
    const float* hp = h + (size_t)node * 128 + head * 16;
    float s = 0.f, d = 0.f;
#pragma unroll
    for (int c = 0; c < 16; ++c) {
        float v = hp[c];
        s += v * a_s[head * 16 + c];
        d += v * a_d[head * 16 + c];
    }
    as_[i] = s; ad_[i] = d;
}
// layer2: one thread per node; h2[N,16], a[1,16]
__global__ void k_alpha2(const float* __restrict__ h2, const float* __restrict__ a_s,
                         const float* __restrict__ a_d, float* as_, float* ad_,
                         int N) {
    int n = blockIdx.x * 256 + threadIdx.x;
    if (n >= N) return;
    float s = 0.f, d = 0.f;
#pragma unroll
    for (int c = 0; c < 16; ++c) {
        float v = h2[(size_t)n * 16 + c];
        s += v * a_s[c];
        d += v * a_d[c];
    }
    as_[n] = s; ad_[n] = d;
}

__device__ __forceinline__ void edge_sd(const int* ei, int E, int e, int& s, int& d) {
    if (e < E) { s = ei[e]; d = ei[E + e]; }
    else       { s = e - E; d = e - E; }      // self loops appended
}
__device__ __forceinline__ float lrelu(float x, float k) { return x > 0.f ? x : k * x; }

// ---- edge passes, layer 1 (thread per edge*head or edge*lane) -------------
__global__ void k_emax1(const int* ei, int E, int Etot,
                        const float* as_, const float* ad_, int* mx, int N) {
    long long i = (long long)blockIdx.x * 256 + threadIdx.x;
    if (i >= (long long)Etot * 8) return;
    int e = (int)(i >> 3), h = (int)(i & 7), s, d;
    edge_sd(ei, E, e, s, d);
    float sc = lrelu(as_[s * 8 + h] + ad_[d * 8 + h], NEG_ATT);
    atomicMax(mx + d * 8 + h, fkey(sc));
}
__global__ void k_esum1(const int* ei, int E, int Etot,
                        const float* as_, const float* ad_,
                        const int* mx, float* dn, int N) {
    long long i = (long long)blockIdx.x * 256 + threadIdx.x;
    if (i >= (long long)Etot * 8) return;
    int e = (int)(i >> 3), h = (int)(i & 7), s, d;
    edge_sd(ei, E, e, s, d);
    float sc = lrelu(as_[s * 8 + h] + ad_[d * 8 + h], NEG_ATT);
    atomicAdd(dn + d * 8 + h, __expf(sc - fval(mx[d * 8 + h])));
}
// 32 lanes per edge; lane handles 4 channels (head = lane/4)
__global__ void k_eagg1(const int* ei, int E, int Etot,
                        const float* as_, const float* ad_,
                        const int* mx, const float* dn,
                        const float* __restrict__ h1, float* agg) {
    long long i = (long long)blockIdx.x * 256 + threadIdx.x;
    if (i >= (long long)Etot * 32) return;
    int e = (int)(i >> 5);
    int lane = (int)(i & 31);
    int h = lane >> 2, ch0 = lane * 4, s, d;
    edge_sd(ei, E, e, s, d);
    float sc = lrelu(as_[s * 8 + h] + ad_[d * 8 + h], NEG_ATT);
    float w  = __expf(sc - fval(mx[d * 8 + h])) / dn[d * 8 + h];
    const float4 hv = *(const float4*)(h1 + (size_t)s * 128 + ch0);
    float* op = agg + (size_t)d * 128 + ch0;
    atomicAdd(op + 0, w * hv.x);
    atomicAdd(op + 1, w * hv.y);
    atomicAdd(op + 2, w * hv.z);
    atomicAdd(op + 3, w * hv.w);
}

// bias + leaky_relu(0.01) + cast to bf16 for layer-2 GEMM
__global__ void k_bias_act_cast(const float* agg, const float* b1,
                                unsigned short* hb, long long n) {
    long long i = (long long)blockIdx.x * 256 + threadIdx.x;
    if (i >= n) return;
    float v = agg[i] + b1[i & 127];
    hb[i] = f2bf(lrelu(v, NEG_ACT));
}

// ---- edge passes, layer 2 (1 head, 16 channels) ---------------------------
__global__ void k_emax2(const int* ei, int E, int Etot,
                        const float* as_, const float* ad_, int* mx) {
    int e = blockIdx.x * 256 + threadIdx.x;
    if (e >= Etot) return;
    int s, d; edge_sd(ei, E, e, s, d);
    atomicMax(mx + d, fkey(lrelu(as_[s] + ad_[d], NEG_ATT)));
}
__global__ void k_esum2(const int* ei, int E, int Etot,
                        const float* as_, const float* ad_,
                        const int* mx, float* dn) {
    int e = blockIdx.x * 256 + threadIdx.x;
    if (e >= Etot) return;
    int s, d; edge_sd(ei, E, e, s, d);
    float sc = lrelu(as_[s] + ad_[d], NEG_ATT);
    atomicAdd(dn + d, __expf(sc - fval(mx[d])));
}
// 16 lanes per edge
__global__ void k_eagg2(const int* ei, int E, int Etot,
                        const float* as_, const float* ad_,
                        const int* mx, const float* dn,
                        const float* __restrict__ h2, float* agg) {
    long long i = (long long)blockIdx.x * 256 + threadIdx.x;
    if (i >= (long long)Etot * 16) return;
    int e = (int)(i >> 4), c = (int)(i & 15), s, d;
    edge_sd(ei, E, e, s, d);
    float sc = lrelu(as_[s] + ad_[d], NEG_ATT);
    float w  = __expf(sc - fval(mx[d])) / dn[d];
    atomicAdd(agg + (size_t)d * 16 + c, w * h2[(size_t)s * 16 + c]);
}

// ---- final bias + row softmax over 16 classes -----------------------------
__global__ void k_softmax(const float* agg, const float* b2, float* out, int N) {
    int n = blockIdx.x * 256 + threadIdx.x;
    if (n >= N) return;
    float v[16], m = -3.0e38f;
#pragma unroll
    for (int c = 0; c < 16; ++c) { v[c] = agg[(size_t)n * 16 + c] + b2[c]; m = fmaxf(m, v[c]); }
    float sum = 0.f;
#pragma unroll
    for (int c = 0; c < 16; ++c) { v[c] = expf(v[c] - m); sum += v[c]; }
    float inv = 1.f / sum;
#pragma unroll
    for (int c = 0; c < 16; ++c) out[(size_t)n * 16 + c] = v[c] * inv;
}

// ---------------------------------------------------------------------------
extern "C" void kernel_launch(void* const* d_in, const int* in_sizes, int n_in,
                              void* d_out, int out_size, void* d_ws, size_t ws_size,
                              hipStream_t stream) {
    const float* x   = (const float*)d_in[0];
    const int*   ei  = (const int*)  d_in[1];
    const float* W1  = (const float*)d_in[2];
    const float* a1s = (const float*)d_in[3];
    const float* a1d = (const float*)d_in[4];
    const float* b1  = (const float*)d_in[5];
    const float* W2  = (const float*)d_in[6];
    const float* a2s = (const float*)d_in[7];
    const float* a2d = (const float*)d_in[8];
    const float* b2  = (const float*)d_in[9];
    float* out = (float*)d_out;

    const int N    = in_sizes[0] / 128;
    const int E    = in_sizes[1] / 2;
    const int Etot = E + N;               // with self loops

    // workspace carve-out (256B aligned)
    char* w = (char*)d_ws;
    auto alloc = [&](size_t bytes) -> void* {
        void* p = (void*)w;
        w += (bytes + 255) & ~(size_t)255;
        return p;
    };
    unsigned short* xb  = (unsigned short*)alloc((size_t)N * 128 * 2);
    unsigned short* w1t = (unsigned short*)alloc(128 * 128 * 2);
    unsigned short* w2t = (unsigned short*)alloc(16 * 128 * 2);
    float* h1   = (float*)alloc((size_t)N * 128 * 4);
    float* as1  = (float*)alloc((size_t)N * 8 * 4);
    float* ad1  = (float*)alloc((size_t)N * 8 * 4);
    int*   mx1  = (int*)  alloc((size_t)N * 8 * 4);
    float* dn1  = (float*)alloc((size_t)N * 8 * 4);
    float* agg1 = (float*)alloc((size_t)N * 128 * 4);
    unsigned short* hb = (unsigned short*)alloc((size_t)N * 128 * 2);
    float* h2   = (float*)alloc((size_t)N * 16 * 4);
    float* as2  = (float*)alloc((size_t)N * 4);
    float* ad2  = (float*)alloc((size_t)N * 4);
    int*   mx2  = (int*)  alloc((size_t)N * 4);
    float* dn2  = (float*)alloc((size_t)N * 4);
    float* agg2 = (float*)alloc((size_t)N * 16 * 4);

    const int B = 256;
    auto g = [](long long n) { return (unsigned)((n + 255) / 256); };

    // casts
    k_cast_bf16<<<g((long long)N * 128), B, 0, stream>>>(x, xb, (long long)N * 128);
    k_cast_w_t <<<g(128 * 128), B, 0, stream>>>(W1, w1t, 128, 128);
    k_cast_w_t <<<g(128 * 16),  B, 0, stream>>>(W2, w2t, 128, 16);

    // init accumulators
    k_fill_i32<<<g((long long)N * 8),  B, 0, stream>>>(mx1, (int)0x80000000, (long long)N * 8);
    k_fill_f32<<<g((long long)N * 8),  B, 0, stream>>>(dn1, 0.f, (long long)N * 8);
    k_fill_f32<<<g((long long)N * 128),B, 0, stream>>>(agg1, 0.f, (long long)N * 128);
    k_fill_i32<<<g(N), B, 0, stream>>>(mx2, (int)0x80000000, N);
    k_fill_f32<<<g(N), B, 0, stream>>>(dn2, 0.f, N);
    k_fill_f32<<<g((long long)N * 16), B, 0, stream>>>(agg2, 0.f, (long long)N * 16);

    // ---- layer 1 ----
    const int ntiles = N / 16;
    k_gemm_wmma<128><<<(ntiles + 7) / 8, B, 0, stream>>>(xb, w1t, h1, ntiles);
    int rem = N - ntiles * 16;
    if (rem > 0)
        k_gemm_tail<<<g((long long)rem * 128), B, 0, stream>>>(xb, w1t, h1, ntiles * 16, rem, 128);

    k_alpha1<<<g((long long)N * 8), B, 0, stream>>>(h1, a1s, a1d, as1, ad1, N);
    k_emax1<<<g((long long)Etot * 8),  B, 0, stream>>>(ei, E, Etot, as1, ad1, mx1, N);
    k_esum1<<<g((long long)Etot * 8),  B, 0, stream>>>(ei, E, Etot, as1, ad1, mx1, dn1, N);
    k_eagg1<<<g((long long)Etot * 32), B, 0, stream>>>(ei, E, Etot, as1, ad1, mx1, dn1, h1, agg1);
    k_bias_act_cast<<<g((long long)N * 128), B, 0, stream>>>(agg1, b1, hb, (long long)N * 128);

    // ---- layer 2 ----
    k_gemm_wmma<16><<<(ntiles + 7) / 8, B, 0, stream>>>(hb, w2t, h2, ntiles);
    if (rem > 0)
        k_gemm_tail<<<g((long long)rem * 16), B, 0, stream>>>(hb, w2t, h2, ntiles * 16, rem, 16);

    k_alpha2<<<g(N), B, 0, stream>>>(h2, a2s, a2d, as2, ad2, N);
    k_emax2<<<g(Etot), B, 0, stream>>>(ei, E, Etot, as2, ad2, mx2);
    k_esum2<<<g(Etot), B, 0, stream>>>(ei, E, Etot, as2, ad2, mx2, dn2);
    k_eagg2<<<g((long long)Etot * 16), B, 0, stream>>>(ei, E, Etot, as2, ad2, mx2, dn2, h2, agg2);

    k_softmax<<<g(N), B, 0, stream>>>(agg2, b2, out, N);
}
